// GroupedQueryAttention_32521492365534
// MI455X (gfx1250) — matrix operand order repaired and checked by reading, not compiled, not run on hardware
//
#include <hip/hip_runtime.h>

// ---------------------------------------------------------------------------
// GQA forward for MI455X (gfx1250): bf16 WMMA everywhere, fp32 accumulate.
// B=2, S=2048, D_MODEL=1024, H=16, KV=4, D=64.  M = B*S = 4096 tokens.
// All LDS tiles are staged with GLOBAL_LOAD_ASYNC_TO_LDS_B128 and
// double-buffered: s_wait_asynccnt with a credit overlaps DMA with WMMA.
// ---------------------------------------------------------------------------

typedef __attribute__((ext_vector_type(4)))  int    v4i;
typedef __attribute__((ext_vector_type(8)))  int    v8i;
typedef __attribute__((ext_vector_type(8)))  float  v8f;
typedef __attribute__((ext_vector_type(16))) __bf16 v16bf;

#define DMODEL 1024
#define NHEAD  16
#define NKV    4
#define DHEAD  64
#define SEQ    2048
#define BATCH  2
#define MTOK   (BATCH * SEQ)               // 4096
#define NQKV   (DMODEL + 2 * NKV * DHEAD)  // 1536

#if defined(__gfx1250__) && __has_builtin(__builtin_amdgcn_global_load_async_to_lds_b128)
#define HAVE_ASYNC 1
#else
#define HAVE_ASYNC 0
#endif

typedef __attribute__((address_space(1))) v4i* gv4i_p;  // global
typedef __attribute__((address_space(3))) v4i* lv4i_p;  // LDS

// copy 16B global -> LDS (async DMA path on CDNA5, tracked by ASYNCcnt)
__device__ __forceinline__ void cp16(const unsigned short* g,
                                     unsigned short* l) {
#if HAVE_ASYNC
  __builtin_amdgcn_global_load_async_to_lds_b128((gv4i_p)g, (lv4i_p)l, 0, 0);
#else
  *(v4i*)l = *(const v4i*)g;
#endif
}

// wait until at most N async ops outstanding (per-wave; in-order completion)
template <int N>
__device__ __forceinline__ void wait_async_le() {
#if HAVE_ASYNC
#if __has_builtin(__builtin_amdgcn_s_wait_asynccnt)
  __builtin_amdgcn_s_wait_asynccnt(N);
#else
  asm volatile("s_wait_asynccnt %0" ::"i"(N) : "memory");
#endif
#endif
}

__device__ __forceinline__ unsigned short f2bf(float f) {
  unsigned u = __float_as_uint(f);
  u += 0x7fffu + ((u >> 16) & 1u);  // round to nearest even
  return (unsigned short)(u >> 16);
}

__device__ __forceinline__ v8f vzero() {
  v8f z;
#pragma unroll
  for (int i = 0; i < 8; i++) z[i] = 0.0f;
  return z;
}

// A-fragment (16x32 bf16) from a row-major tile. ISA 7.12.2:
// lane m = L&15; half g = L>>4; dwords 0-3 hold K = g*8+[0..7],
// dwords 4-7 hold K = 16+g*8+[0..7].  Two contiguous 16B loads.
__device__ __forceinline__ v16bf ldA(const unsigned short* t, int stride,
                                     int lane, int c0) {
  int m = lane & 15, g = lane >> 4;
  const unsigned short* r = t + (size_t)m * stride + c0;
  union { v4i p[2]; v16bf b; } u;
  u.p[0] = *(const v4i*)(r + g * 8);
  u.p[1] = *(const v4i*)(r + 16 + g * 8);
  return u.b;
}

// B-fragment (32x16 bf16) from tile[n][k]: lane n = L&15, half g = L>>4,
// dwords 0-3 hold K = g*8+[0..7], dwords 4-7 hold K = 16+g*8+[0..7].
__device__ __forceinline__ v16bf ldB(const unsigned short* t, int stride,
                                     int lane, int k0) {
  int n = lane & 15, g = lane >> 4;
  const unsigned short* r = t + (size_t)n * stride + k0;
  union { v4i p[2]; v16bf b; } u;
  u.p[0] = *(const v4i*)(r + g * 8);
  u.p[1] = *(const v4i*)(r + 16 + g * 8);
  return u.b;
}

// ---------------------------------------------------------------------------
// Packing kernels: fp32 -> bf16, weights stored TRANSPOSED (N x K)
// ---------------------------------------------------------------------------
__global__ void cvt_f32_bf16(const float* __restrict__ src,
                             unsigned short* __restrict__ dst, int n) {
  int i = blockIdx.x * blockDim.x + threadIdx.x;
  if (i < n) dst[i] = f2bf(src[i]);
}

// wqkvT[n][k], n in [0,1536): cols of [Wq | Wk | Wv]
__global__ void pack_wqkvT(const float* __restrict__ Wq,
                           const float* __restrict__ Wk,
                           const float* __restrict__ Wv,
                           unsigned short* __restrict__ outT) {
  int i = blockIdx.x * blockDim.x + threadIdx.x;  // over 1536*1024
  int n = i / DMODEL, k = i % DMODEL;
  float v;
  if (n < DMODEL)            v = Wq[(size_t)k * DMODEL + n];
  else if (n < DMODEL + 256) v = Wk[(size_t)k * 256 + (n - DMODEL)];
  else                       v = Wv[(size_t)k * 256 + (n - DMODEL - 256)];
  outT[i] = f2bf(v);
}

// woT[n][k] = Wo[k][n]
__global__ void pack_woT(const float* __restrict__ Wo,
                         unsigned short* __restrict__ outT) {
  int i = blockIdx.x * blockDim.x + threadIdx.x;  // over 1024*1024
  int n = i / DMODEL, k = i % DMODEL;
  outT[i] = f2bf(Wo[(size_t)k * DMODEL + n]);
}

// ---------------------------------------------------------------------------
// GEMM: C(MxN,f32) = A(MxK,bf16 row-major) * BT(NxK,bf16 row-major)
// block tile 128x64, K-step 64, 8 waves (4 x 2), wave tile 32x32 = 2x2 WMMA.
// Double-buffered async staging: 6 async ops per thread per stage.
// ---------------------------------------------------------------------------
#define GTS 72  // LDS row stride in halfwords (144B: 16B aligned, bank spread)

__global__ __launch_bounds__(256) void gemm_bf16(
    const unsigned short* __restrict__ A, const unsigned short* __restrict__ BT,
    float* __restrict__ C, int M, int N, int K) {
  __shared__ unsigned short sA[2][128 * GTS];
  __shared__ unsigned short sB[2][64 * GTS];  // sB[p][n][k]

  int tid = threadIdx.x;
  int lane = tid & 31;
  int wave = tid >> 5;
  int wm = wave & 3;   // rows wm*32
  int wn = wave >> 2;  // cols wn*32
  int bm = blockIdx.x * 128;
  int bn = blockIdx.y * 64;

  auto stage = [&](int kt, int p) {
    // A tile: 128 rows x 64 hw; 2 threads/row, 32 hw (4x16B) each
    int r = tid >> 1, c = (tid & 1) * 32;
    const unsigned short* src = A + (size_t)(bm + r) * K + kt * 64 + c;
    unsigned short* dst = sA[p] + r * GTS + c;
#pragma unroll
    for (int i = 0; i < 4; i++) cp16(src + i * 8, dst + i * 8);
    // BT tile: 64 rows x 64 hw; 4 threads/row, 16 hw (2x16B) each
    int r2 = tid >> 2, c2 = (tid & 3) * 16;
    const unsigned short* src2 = BT + (size_t)(bn + r2) * K + kt * 64 + c2;
    unsigned short* dst2 = sB[p] + r2 * GTS + c2;
    cp16(src2, dst2);
    cp16(src2 + 8, dst2 + 8);
  };

  v8f acc[2][2];
#pragma unroll
  for (int i = 0; i < 2; i++)
#pragma unroll
    for (int j = 0; j < 2; j++) acc[i][j] = vzero();

  int nk = K >> 6;
  stage(0, 0);
  for (int kt = 0; kt < nk; kt++) {
    int p = kt & 1;
    bool more = (kt + 1 < nk);
    if (more) stage(kt + 1, p ^ 1);        // DMA next tile behind compute
    if (more) wait_async_le<6>();          // current tile resident
    else      wait_async_le<0>();
    __syncthreads();

#pragma unroll
    for (int ks = 0; ks < 64; ks += 32) {
      v16bf a0 = ldA(sA[p] + (wm * 32) * GTS, GTS, lane, ks);
      v16bf a1 = ldA(sA[p] + (wm * 32 + 16) * GTS, GTS, lane, ks);
      v16bf b0 = ldB(sB[p] + (wn * 32) * GTS, GTS, lane, ks);
      v16bf b1 = ldB(sB[p] + (wn * 32 + 16) * GTS, GTS, lane, ks);
      acc[0][0] = __builtin_amdgcn_wmma_f32_16x16x32_bf16(
          false, a0, false, b0, (short)0, acc[0][0], false, false);
      acc[0][1] = __builtin_amdgcn_wmma_f32_16x16x32_bf16(
          false, a0, false, b1, (short)0, acc[0][1], false, false);
      acc[1][0] = __builtin_amdgcn_wmma_f32_16x16x32_bf16(
          false, a1, false, b0, (short)0, acc[1][0], false, false);
      acc[1][1] = __builtin_amdgcn_wmma_f32_16x16x32_bf16(
          false, a1, false, b1, (short)0, acc[1][1], false, false);
    }
    __syncthreads();  // all waves done reading buf p before it is re-staged
  }

  // C-layout: vgpr r -> rows r / r+8 by lane half; col = lane&15
  int g = lane >> 4;
  int n0 = bn + wn * 32 + (lane & 15);
#pragma unroll
  for (int i = 0; i < 2; i++)
#pragma unroll
    for (int j = 0; j < 2; j++)
#pragma unroll
      for (int r = 0; r < 8; r++) {
        int row = bm + wm * 32 + i * 16 + r + 8 * g;
        C[(size_t)row * N + n0 + j * 16] = acc[i][j][r];
      }
}

// ---------------------------------------------------------------------------
// RoPE + repack:  yqkv (4096 x 1536 f32) ->
//   qbf (b,h,s,d) bf16 ; kbf (b,kv,s,d) bf16 ; vT (b,kv,d,s) bf16
// ---------------------------------------------------------------------------
__global__ __launch_bounds__(256) void rope_repack(
    const float* __restrict__ y, unsigned short* __restrict__ qbf,
    unsigned short* __restrict__ kbf, unsigned short* __restrict__ vT) {
  int t = blockIdx.x;  // token
  int b = t >> 11, s = t & 2047;
  const float LOG1E4 = 9.2103403720f;  // ln(10000)
  for (int idx = threadIdx.x; idx < NQKV; idx += 256) {
    float val = y[(size_t)t * NQKV + idx];
    if (idx < DMODEL) {  // Q with RoPE
      int h = idx >> 6, d = idx & 63;
      float ang = (float)s * __expf(-(float)(d & 31) * (LOG1E4 / 32.0f));
      const float* base = y + (size_t)t * NQKV + h * 64;
      float rot = (d < 32) ? -base[d + 32] : base[d - 32];
      float o = val * __cosf(ang) + rot * __sinf(ang);
      qbf[(((size_t)(b * NHEAD + h)) * SEQ + s) * DHEAD + d] = f2bf(o);
    } else if (idx < DMODEL + NKV * DHEAD) {  // K with RoPE
      int j = idx - DMODEL;
      int kv = j >> 6, d = j & 63;
      float ang = (float)s * __expf(-(float)(d & 31) * (LOG1E4 / 32.0f));
      const float* base = y + (size_t)t * NQKV + DMODEL + kv * 64;
      float rot = (d < 32) ? -base[d + 32] : base[d - 32];
      float o = val * __cosf(ang) + rot * __sinf(ang);
      kbf[(((size_t)(b * NKV + kv)) * SEQ + s) * DHEAD + d] = f2bf(o);
    } else {  // V transposed
      int j = idx - DMODEL - NKV * DHEAD;
      int kv = j >> 6, d = j & 63;
      vT[(((size_t)(b * NKV + kv)) * DHEAD + d) * SEQ + s] = f2bf(val);
    }
  }
}

// ---------------------------------------------------------------------------
// Flash attention: grid (S/64, B*H), 128 threads = 4 waves.
// Each wave owns 16 query rows; double-buffered 64-key K/V tiles in LDS
// (8 async ops per thread per stage).
// ---------------------------------------------------------------------------
#define FTS 72

__global__ __launch_bounds__(128) void fa_kernel(
    const unsigned short* __restrict__ qbf,
    const unsigned short* __restrict__ kbf,
    const unsigned short* __restrict__ vT,
    unsigned short* __restrict__ ctx) {
  __shared__ unsigned short sK[2][64 * FTS];  // [p][key][d]
  __shared__ unsigned short sV[2][64 * FTS];  // [p][d][key]
  __shared__ unsigned short sP[4][16 * FTS];  // per-wave probs [qrow][key]

  int tid = threadIdx.x, lane = tid & 31, wave = tid >> 5;
  int g = lane >> 4;
  int q0 = blockIdx.x * 64;
  int bh = blockIdx.y;
  int b = bh >> 4, h = bh & 15;
  int kv = h >> 2;  // 4 q-heads per kv-head

  const unsigned short* qhead = qbf + ((size_t)bh * SEQ) * DHEAD;
  const unsigned short* khead = kbf + ((size_t)(b * NKV + kv) * SEQ) * DHEAD;
  const unsigned short* vhead = vT + ((size_t)(b * NKV + kv) * DHEAD) * SEQ;

  auto stage = [&](int kt, int p) {
    int r = tid >> 1, c = (tid & 1) * 32;
    const unsigned short* srck = khead + (size_t)(kt * 64 + r) * DHEAD + c;
    unsigned short* dstk = sK[p] + r * FTS + c;
    const unsigned short* srcv = vhead + (size_t)r * SEQ + kt * 64 + c;
    unsigned short* dstv = sV[p] + r * FTS + c;
#pragma unroll
    for (int i = 0; i < 4; i++) cp16(srck + i * 8, dstk + i * 8);
#pragma unroll
    for (int i = 0; i < 4; i++) cp16(srcv + i * 8, dstv + i * 8);
  };

  int qrow = q0 + wave * 16;
  // Q A-fragments live in registers for the whole loop
  v16bf qa0 = ldA(qhead + (size_t)qrow * DHEAD, DHEAD, lane, 0);
  v16bf qa1 = ldA(qhead + (size_t)qrow * DHEAD, DHEAD, lane, 32);

  v8f acc[4];
#pragma unroll
  for (int nt = 0; nt < 4; nt++) acc[nt] = vzero();
  float mi[8], li[8];
#pragma unroll
  for (int r = 0; r < 8; r++) { mi[r] = -1e30f; li[r] = 0.0f; }
  const float scale = 0.125f;  // 1/sqrt(64)

  int ntiles = blockIdx.x + 1;  // causal: keys [0, q0+64)
  stage(0, 0);
  for (int kt = 0; kt < ntiles; kt++) {
    int p = kt & 1;
    int k0 = kt * 64;
    bool more = (kt + 1 < ntiles);
    if (more) stage(kt + 1, p ^ 1);
    if (more) wait_async_le<8>();
    else      wait_async_le<0>();
    __syncthreads();

    // scores: 16q x 64k
    v8f s[4];
#pragma unroll
    for (int nt = 0; nt < 4; nt++) {
      v16bf kb0 = ldB(sK[p] + (nt * 16) * FTS, FTS, lane, 0);
      v16bf kb1 = ldB(sK[p] + (nt * 16) * FTS, FTS, lane, 32);
      v8f z = vzero();
      z = __builtin_amdgcn_wmma_f32_16x16x32_bf16(false, qa0, false, kb0,
                                                  (short)0, z, false, false);
      s[nt] = __builtin_amdgcn_wmma_f32_16x16x32_bf16(
          false, qa1, false, kb1, (short)0, z, false, false);
    }

    bool need_mask = (k0 + 63 > qrow);
    // online softmax per accumulator row r (lane half g owns row r + 8*g)
#pragma unroll
    for (int r = 0; r < 8; r++) {
      int row = qrow + r + 8 * g;
      float mx = -1e30f;
#pragma unroll
      for (int nt = 0; nt < 4; nt++) {
        float v = s[nt][r] * scale;
        if (need_mask) {
          int key = k0 + nt * 16 + (lane & 15);
          if (key > row) v = -1e30f;
        }
        s[nt][r] = v;
        mx = fmaxf(mx, v);
      }
#pragma unroll
      for (int off = 8; off; off >>= 1) mx = fmaxf(mx, __shfl_xor(mx, off, 16));
      float mnew = fmaxf(mi[r], mx);
      float corr = __expf(mi[r] - mnew);
      float rs = 0.0f;
#pragma unroll
      for (int nt = 0; nt < 4; nt++) {
        float pe = __expf(s[nt][r] - mnew);
        s[nt][r] = pe;
        rs += pe;
      }
#pragma unroll
      for (int off = 8; off; off >>= 1) rs += __shfl_xor(rs, off, 16);
      li[r] = li[r] * corr + rs;
      mi[r] = mnew;
#pragma unroll
      for (int nt = 0; nt < 4; nt++) acc[nt][r] *= corr;
    }

    // P: C-layout -> per-wave LDS -> A-fragments (same-wave DS ops in order)
    unsigned short* pw = sP[wave];
#pragma unroll
    for (int nt = 0; nt < 4; nt++)
#pragma unroll
      for (int r = 0; r < 8; r++)
        pw[(r + 8 * g) * FTS + nt * 16 + (lane & 15)] = f2bf(s[nt][r]);

#pragma unroll
    for (int ks = 0; ks < 2; ks++) {
      v16bf pa = ldA(pw, FTS, lane, ks * 32);
#pragma unroll
      for (int nt = 0; nt < 4; nt++) {
        v16bf vb = ldB(sV[p] + (nt * 16) * FTS, FTS, lane, ks * 32);
        acc[nt] = __builtin_amdgcn_wmma_f32_16x16x32_bf16(
            false, pa, false, vb, (short)0, acc[nt], false, false);
      }
    }
    __syncthreads();  // everyone done with buf p before it is re-staged
  }

  // epilogue: ctx[b][s][h*64+d] = acc/l  (bf16, row-major 4096x1024)
#pragma unroll
  for (int nt = 0; nt < 4; nt++)
#pragma unroll
    for (int r = 0; r < 8; r++) {
      int srow = qrow + r + 8 * g;
      float o = acc[nt][r] / li[r];
      size_t tok = (size_t)b * SEQ + srow;
      ctx[tok * DMODEL + h * 64 + nt * 16 + (lane & 15)] = f2bf(o);
    }
}

// ---------------------------------------------------------------------------
// launch
// ---------------------------------------------------------------------------
extern "C" void kernel_launch(void* const* d_in, const int* in_sizes, int n_in,
                              void* d_out, int out_size, void* d_ws,
                              size_t ws_size, hipStream_t stream) {
  const float* x  = (const float*)d_in[0];
  const float* Wq = (const float*)d_in[1];
  const float* Wk = (const float*)d_in[2];
  const float* Wv = (const float*)d_in[3];
  const float* Wo = (const float*)d_in[4];
  float* out = (float*)d_out;

  char* ws = (char*)d_ws;
  unsigned short* xb    = (unsigned short*)ws;  ws += (size_t)MTOK * DMODEL * 2;
  unsigned short* wqkvT = (unsigned short*)ws;  ws += (size_t)DMODEL * NQKV * 2;
  unsigned short* woT   = (unsigned short*)ws;  ws += (size_t)DMODEL * DMODEL * 2;
  float*          yqkv  = (float*)ws;           ws += (size_t)MTOK * NQKV * 4;
  unsigned short* qbf   = (unsigned short*)ws;  ws += (size_t)BATCH * NHEAD * SEQ * DHEAD * 2;
  unsigned short* kbf   = (unsigned short*)ws;  ws += (size_t)BATCH * NKV * SEQ * DHEAD * 2;
  unsigned short* vT    = (unsigned short*)ws;  ws += (size_t)BATCH * NKV * SEQ * DHEAD * 2;
  unsigned short* ctx   = (unsigned short*)ws;  ws += (size_t)MTOK * DMODEL * 2;

  // 1. convert / transpose-pack to bf16
  cvt_f32_bf16<<<(MTOK * DMODEL) / 256, 256, 0, stream>>>(x, xb, MTOK * DMODEL);
  pack_wqkvT<<<(DMODEL * NQKV) / 256, 256, 0, stream>>>(Wq, Wk, Wv, wqkvT);
  pack_woT<<<(DMODEL * DMODEL) / 256, 256, 0, stream>>>(Wo, woT);
  // 2. fused QKV projection
  gemm_bf16<<<dim3(MTOK / 128, NQKV / 64), 256, 0, stream>>>(
      xb, wqkvT, yqkv, MTOK, NQKV, DMODEL);
  // 3. RoPE + head-major repack (+ V transpose)
  rope_repack<<<MTOK, 256, 0, stream>>>(yqkv, qbf, kbf, vT);
  // 4. causal flash attention
  fa_kernel<<<dim3(SEQ / 64, BATCH * NHEAD), 128, 0, stream>>>(qbf, kbf, vT,
                                                               ctx);
  // 5. output projection -> fp32 out
  gemm_bf16<<<dim3(MTOK / 128, DMODEL / 64), 256, 0, stream>>>(
      ctx, woT, out, MTOK, DMODEL, DMODEL);
}
